// AttnDecoderRNN_50946902065219
// MI455X (gfx1250) — compile-verified, matrix operand order checked by reading, not run on hardware
//
#include <hip/hip_runtime.h>
#include <cstddef>
#include <cstdint>

// ---------------- problem dims ----------------
constexpr int Bn = 64, NP = 49, ART = 180, En = 512, Hn = 512, An = 512;
constexpr int Vn = 30000, TMAXn = 25, Lc = ART + NP; // Lc = 229
constexpr int G4 = 4 * Hn;                           // 2048
constexpr int LDT = 40;                              // padded LDS row (bf16 elems): 80B, conflict-free & 16B aligned

// ---------------- types ----------------
typedef __attribute__((ext_vector_type(8)))  __bf16 v8bf;
typedef __attribute__((ext_vector_type(16))) __bf16 v16bf;
typedef __attribute__((ext_vector_type(8)))  float  v8f;

__device__ __forceinline__ unsigned short f2bf(float f) {
  unsigned int u = __float_as_uint(f);
  u += 0x7FFFu + ((u >> 16) & 1u);   // round-to-nearest-even
  return (unsigned short)(u >> 16);
}
__device__ __forceinline__ float bf2f(unsigned short h) {
  return __uint_as_float(((unsigned int)h) << 16);
}
__device__ __forceinline__ float sigm(float x) { return 1.f / (1.f + expf(-x)); }

// ---------------- WMMA fragment load from an LDS tile ----------------
// 16-bit A-matrix 16x32 layout (ISA 7.12.2): lanes 0-15 hold K{0..7,16..23},
// lanes 16-31 hold K{8..15,24..31}, row = lane&15. Same pattern serves B
// fragments because weights are stored [N,K] row-major (x @ W^T).
__device__ __forceinline__ v16bf lds_frag(const unsigned short* tile, int row0) {
  int lane = threadIdx.x & 31;
  int r = row0 + (lane & 15);
  int kk = (lane & 16) >> 1;  // +8 for upper half-wave
  const __bf16* p = (const __bf16*)tile + r * LDT + kk;
  v8bf lo = *(const v8bf*)p;
  v8bf hi = *(const v8bf*)(p + 16);
  v16bf f;
#pragma unroll
  for (int i = 0; i < 8; ++i) { f[i] = lo[i]; f[i + 8] = hi[i]; }
  return f;
}

__device__ __forceinline__ void compute_tile(const unsigned short* As, const unsigned short* Bs,
                                             int wm, int wn, v8f acc[4]) {
  v16bf af = lds_frag(As, wm);
#pragma unroll
  for (int j = 0; j < 4; ++j) {
    v16bf wf = lds_frag(Bs, wn + j * 16);
    acc[j] = __builtin_amdgcn_wmma_f32_16x16x32_bf16(
        false, af, false, wf, (short)0, acc[j], false, false);
  }
}

// ---------------- WMMA GEMM: C[M,N] = A[M,K] @ W[N,K]^T ----------------
// Block = 256 threads = 8 wave32 waves; block tile 64(M) x 128(N), K-step 32.
// LDS double-buffered staging (30KB/block out of 320KB/WGP): each K-step a
// block fetches A(4KB)+B(8KB) from global exactly once, while the next tile's
// global loads are in flight during the current tile's WMMAs.
template <bool BIAS, bool ACCUM, bool RELU, bool OUTBF16>
__global__ __launch_bounds__(256) void k_gemm(
    const unsigned short* __restrict__ Abf,
    const unsigned short* __restrict__ Wbf,
    const float* __restrict__ bias,
    float* __restrict__ Cf,
    unsigned short* __restrict__ Cbf,
    int M, int N, int K) {
  __shared__ __align__(16) unsigned short Asm[2][64 * LDT];
  __shared__ __align__(16) unsigned short Bsm[2][128 * LDT];

  int tid = threadIdx.x;
  int wave = tid >> 5, lane = tid & 31;
  int m_blk = blockIdx.y * 64, n_blk = blockIdx.x * 128;
  int wm = (wave & 3) * 16;   // wave's M offset within tile
  int wn = (wave >> 2) * 64;  // wave's N offset within tile

  // staging: each thread moves one A b128 and two B b128 per K-step
  int srow = tid >> 2;          // 0..63
  int schunk = (tid & 3) * 8;   // bf16-elem offset {0,8,16,24}
  int arow = m_blk + srow;          if (arow > M - 1) arow = M - 1;
  int brow0 = n_blk + srow;         if (brow0 > N - 1) brow0 = N - 1;
  int brow1 = n_blk + srow + 64;    if (brow1 > N - 1) brow1 = N - 1;
  const __bf16* aG  = (const __bf16*)Abf + (size_t)arow  * K + schunk;
  const __bf16* b0G = (const __bf16*)Wbf + (size_t)brow0 * K + schunk;
  const __bf16* b1G = (const __bf16*)Wbf + (size_t)brow1 * K + schunk;
  __bf16* aL0 = (__bf16*)&Asm[0][srow * LDT + schunk];
  __bf16* bL0 = (__bf16*)&Bsm[0][srow * LDT + schunk];
  __bf16* bL1 = (__bf16*)&Bsm[0][(srow + 64) * LDT + schunk];
  const int As_stride = 64 * LDT, Bs_stride = 128 * LDT;

  v8f acc[4];
#pragma unroll
  for (int j = 0; j < 4; ++j)
#pragma unroll
    for (int i = 0; i < 8; ++i) acc[j][i] = 0.f;

  // prologue: stage K-step 0 into buffer 0
  {
    v8bf ra  = *(const v8bf*)aG;
    v8bf rb0 = *(const v8bf*)b0G;
    v8bf rb1 = *(const v8bf*)b1G;
    *(v8bf*)aL0 = ra;
    *(v8bf*)bL0 = rb0;
    *(v8bf*)bL1 = rb1;
  }
  __syncthreads();

  int nsteps = K >> 5;
  for (int s = 0; s < nsteps - 1; ++s) {
    int cur = s & 1, nxt = cur ^ 1;
    int k0 = (s + 1) << 5;
    // issue next tile's global loads; latency overlaps the WMMAs below
    v8bf ra  = *(const v8bf*)(aG + k0);
    v8bf rb0 = *(const v8bf*)(b0G + k0);
    v8bf rb1 = *(const v8bf*)(b1G + k0);

    compute_tile(Asm[cur], Bsm[cur], wm, wn, acc);

    __syncthreads();  // everyone done reading LDS before overwrite
    *(v8bf*)(aL0 + nxt * As_stride) = ra;
    *(v8bf*)(bL0 + nxt * Bs_stride) = rb0;
    *(v8bf*)(bL1 + nxt * Bs_stride) = rb1;
    __syncthreads();  // stores visible
  }
  {
    int cur = (nsteps - 1) & 1;
    compute_tile(Asm[cur], Bsm[cur], wm, wn, acc);
  }

#pragma unroll
  for (int j = 0; j < 4; ++j) {
    int n = n_blk + wn + j * 16 + (lane & 15);
    if (n >= N) continue;
#pragma unroll
    for (int r = 0; r < 8; ++r) {
      int m = m_blk + wm + r + ((lane & 16) >> 1);  // lanes>=16 hold M=r+8
      size_t idx = (size_t)m * N + n;
      float v = acc[j][r];
      if (ACCUM) v += Cf[idx];
      if (BIAS) v += bias[n];
      if (RELU) v = fmaxf(v, 0.f);
      if (OUTBF16) Cbf[idx] = f2bf(v); else Cf[idx] = v;
    }
  }
}

// ---------------- elementwise / helper kernels ----------------
__global__ void k_f2bf(const float* __restrict__ s, unsigned short* __restrict__ d, int n) {
  int i = blockIdx.x * 256 + threadIdx.x;
  if (i < n) d[i] = f2bf(s[i]);
}
__global__ void k_zero_f32(float* p, int n) {
  int i = blockIdx.x * 256 + threadIdx.x;
  if (i < n) p[i] = 0.f;
}
__global__ void k_zero_u16(unsigned short* p, int n) {
  int i = blockIdx.x * 256 + threadIdx.x;
  if (i < n) p[i] = 0;
}

// article embeddings -> bf16, time-major [t][b][e]
__global__ void k_gather_art(const int* __restrict__ art, const float* __restrict__ emb1,
                             unsigned short* __restrict__ out) {
  int i = blockIdx.x * 256 + threadIdx.x;
  if (i >= ART * Bn * En) return;
  int e = i & 511;
  int r = i >> 9;            // t*Bn + b
  int b = r % Bn, t = r / Bn;
  int tok = art[b * ART + t];
  out[i] = f2bf(emb1[(size_t)tok * En + e]);
}

// encoder LSTM pointwise: c,h update from gates; writes bf16 h (state) + hs slice
__global__ void k_lstm_step_enc(const float* __restrict__ g, float* __restrict__ c,
                                unsigned short* __restrict__ h_bf,
                                unsigned short* __restrict__ hs) {
  int i = blockIdx.x * 256 + threadIdx.x;
  if (i >= Bn * Hn) return;
  int b = i >> 9, j = i & 511;
  const float* gr = g + (size_t)b * G4;
  float gi = gr[j], gf = gr[Hn + j], gg = gr[2 * Hn + j], go = gr[3 * Hn + j];
  float cn = sigm(gf) * c[i] + sigm(gi) * tanhf(gg);
  float hn = sigm(go) * tanhf(cn);
  c[i] = cn;
  unsigned short hb = f2bf(hn);
  h_bf[i] = hb;
  hs[i] = hb;
}

// concat fwd/bwd hidden states -> [b*ART + t][1024]
__global__ void k_concat_fb(const unsigned short* __restrict__ hsF,
                            const unsigned short* __restrict__ hsB,
                            unsigned short* __restrict__ X2) {
  int i = blockIdx.x * 256 + threadIdx.x;
  if (i >= Bn * ART * 1024) return;
  int j = i & 1023;
  int row = i >> 10;         // b*ART + t
  int t = row % ART, b = row / ART;
  X2[i] = (j < Hn) ? hsF[((size_t)t * Bn + b) * Hn + j]
                   : hsB[((size_t)t * Bn + b) * Hn + (j - Hn)];
}

// hiddens[b][l][h] = (l<ART)? hiddens1 : enc1
__global__ void k_assemble_hiddens(const unsigned short* __restrict__ h1,
                                   const unsigned short* __restrict__ enc1,
                                   unsigned short* __restrict__ hid) {
  int i = blockIdx.x * 256 + threadIdx.x;
  if (i >= Bn * Lc * Hn) return;
  int j = i & 511;
  int r = i >> 9;            // b*Lc + l
  int l = r % Lc, b = r / Lc;
  hid[i] = (l < ART) ? h1[((size_t)b * ART + l) * Hn + j]
                     : enc1[((size_t)b * NP + (l - ART)) * Hn + j];
}

// e[b][l] = full_b + sum_a fullW[a]*tanh(att1[b][l][a] + att2[b][a]); wave per row
__global__ void k_att_scores(const float* __restrict__ att1, const float* __restrict__ att2,
                             const float* __restrict__ fW, const float* __restrict__ fb,
                             float* __restrict__ e) {
  int wave = (blockIdx.x * blockDim.x + threadIdx.x) >> 5;
  int lane = threadIdx.x & 31;
  if (wave >= Bn * Lc) return;
  int b = wave / Lc;
  const float* a1 = att1 + (size_t)wave * An;
  const float* a2 = att2 + (size_t)b * An;
  float s = 0.f;
  for (int a = lane; a < An; a += 32) s += fW[a] * tanhf(a1[a] + a2[a]);
#pragma unroll
  for (int off = 16; off > 0; off >>= 1) s += __shfl_xor(s, off, 32);
  if (lane == 0) e[wave] = s + fb[0];
}

__global__ void k_softmax_L(const float* __restrict__ e, float* __restrict__ alpha) {
  __shared__ float red[256];
  int b = blockIdx.x, tid = threadIdx.x;
  const float* row = e + (size_t)b * Lc;
  float m = -1e30f;
  for (int l = tid; l < Lc; l += 256) m = fmaxf(m, row[l]);
  red[tid] = m; __syncthreads();
  for (int s = 128; s > 0; s >>= 1) { if (tid < s) red[tid] = fmaxf(red[tid], red[tid + s]); __syncthreads(); }
  m = red[0]; __syncthreads();
  float sum = 0.f;
  for (int l = tid; l < Lc; l += 256) sum += expf(row[l] - m);
  red[tid] = sum; __syncthreads();
  for (int s = 128; s > 0; s >>= 1) { if (tid < s) red[tid] += red[tid + s]; __syncthreads(); }
  float inv = 1.f / red[0];
  for (int l = tid; l < Lc; l += 256) alpha[(size_t)b * Lc + l] = expf(row[l] - m) * inv;
}

// awe[b][h] = sum_l alpha[b][l]*hiddens[b][l][h] -> x_bf[b][0..512)
__global__ void k_awe_x(const float* __restrict__ alpha, const unsigned short* __restrict__ hid,
                        unsigned short* __restrict__ x) {
  int b = blockIdx.x, tid = threadIdx.x;
  const float* al = alpha + (size_t)b * Lc;
  for (int h = tid; h < Hn; h += 256) {
    const unsigned short* hb = hid + (size_t)b * Lc * Hn + h;
    float s = 0.f;
    for (int l = 0; l < Lc; ++l) s += al[l] * bf2f(hb[(size_t)l * Hn]);
    x[(size_t)b * 1024 + h] = f2bf(s);
  }
}

// caption embedding for step t -> x_bf[b][512..1024)
__global__ void k_emb_x(const int* __restrict__ caps, const float* __restrict__ embw,
                        unsigned short* __restrict__ x, int t) {
  int i = blockIdx.x * 256 + threadIdx.x;
  if (i >= Bn * En) return;
  int b = i >> 9, j = i & 511;
  int tok = caps[b * (TMAXn + 1) + t];
  x[(size_t)b * 1024 + 512 + j] = f2bf(embw[(size_t)tok * En + j]);
}

// decoder LSTMCell pointwise with active mask
__global__ void k_cell_step(const float* __restrict__ g, const int* __restrict__ clen,
                            float* __restrict__ h, float* __restrict__ c,
                            unsigned short* __restrict__ h_bf, int t) {
  int i = blockIdx.x * 256 + threadIdx.x;
  if (i >= Bn * Hn) return;
  int b = i >> 9, j = i & 511;
  bool active = t < (clen[b] - 1);
  const float* gr = g + (size_t)b * G4;
  float gi = gr[j], gf = gr[Hn + j], gg = gr[2 * Hn + j], go = gr[3 * Hn + j];
  float cn = sigm(gf) * c[i] + sigm(gi) * tanhf(gg);
  float hn = sigm(go) * tanhf(cn);
  if (active) { c[i] = cn; h[i] = hn; }
  h_bf[i] = f2bf(h[i]);
}

// log_softmax over V, masked write into predictions[b][t][:]
__global__ void k_log_softmax(const float* __restrict__ logits, const int* __restrict__ clen,
                              float* __restrict__ preds, int t) {
  __shared__ float red[256];
  int b = blockIdx.x, tid = threadIdx.x;
  const float* row = logits + (size_t)b * Vn;
  bool active = t < (clen[b] - 1);
  float m = -1e30f;
  for (int v = tid; v < Vn; v += 256) m = fmaxf(m, row[v]);
  red[tid] = m; __syncthreads();
  for (int s = 128; s > 0; s >>= 1) { if (tid < s) red[tid] = fmaxf(red[tid], red[tid + s]); __syncthreads(); }
  m = red[0]; __syncthreads();
  float sum = 0.f;
  for (int v = tid; v < Vn; v += 256) sum += expf(row[v] - m);
  red[tid] = sum; __syncthreads();
  for (int s = 128; s > 0; s >>= 1) { if (tid < s) red[tid] += red[tid + s]; __syncthreads(); }
  float lse = m + logf(red[0]);
  float* o = preds + ((size_t)b * TMAXn + t) * Vn;
  for (int v = tid; v < Vn; v += 256) o[v] = active ? (row[v] - lse) : 0.f;
}

// aux outputs: encoded_captions (as float) + decode_lengths
__global__ void k_misc_out(const int* __restrict__ caps, const int* __restrict__ clen,
                           float* __restrict__ out_caps, float* __restrict__ out_dlen) {
  int i = blockIdx.x * 256 + threadIdx.x;
  if (i < Bn * (TMAXn + 1)) out_caps[i] = (float)caps[i];
  if (i < Bn) out_dlen[i] = (float)(clen[i] - 1);
}

// ---------------- host orchestration ----------------
#define GEMM_CALL(BI, AC, RL, OB, A_, W_, BIAS_, CF_, CB_, M_, N_, K_)                     \
  k_gemm<BI, AC, RL, OB><<<dim3(((N_) + 127) / 128, (M_) / 64), 256, 0, stream>>>(         \
      (const unsigned short*)(A_), (const unsigned short*)(W_), (const float*)(BIAS_),     \
      (float*)(CF_), (unsigned short*)(CB_), (M_), (N_), (K_))

extern "C" void kernel_launch(void* const* d_in, const int* in_sizes, int n_in,
                              void* d_out, int out_size, void* d_ws, size_t ws_size,
                              hipStream_t stream) {
  (void)in_sizes; (void)n_in; (void)out_size; (void)ws_size;
  const float* encoder_out = (const float*)d_in[0];
  const float* emb_w   = (const float*)d_in[1];
  const float* emb1_w  = (const float*)d_in[2];
  const float* WihF    = (const float*)d_in[3];
  const float* WhhF    = (const float*)d_in[4];
  const float* bihF    = (const float*)d_in[5];
  const float* bhhF    = (const float*)d_in[6];
  const float* WihB    = (const float*)d_in[7];
  const float* WhhB    = (const float*)d_in[8];
  const float* bihB    = (const float*)d_in[9];
  const float* bhhB    = (const float*)d_in[10];
  const float* f4W     = (const float*)d_in[11];
  const float* f4b     = (const float*)d_in[12];
  const float* imgW    = (const float*)d_in[13];
  const float* imgb    = (const float*)d_in[14];
  const float* encAttW = (const float*)d_in[15];
  const float* encAttb = (const float*)d_in[16];
  const float* decAttW = (const float*)d_in[17];
  const float* decAttb = (const float*)d_in[18];
  const float* fullW   = (const float*)d_in[19];
  const float* fullb   = (const float*)d_in[20];
  const float* cWih    = (const float*)d_in[21];
  const float* cWhh    = (const float*)d_in[22];
  const float* cbih    = (const float*)d_in[23];
  const float* cbhh    = (const float*)d_in[24];
  const float* fcW     = (const float*)d_in[25];
  const float* fcb     = (const float*)d_in[26];
  const int*   caps    = (const int*)d_in[27];
  const int*   arts    = (const int*)d_in[28];
  const int*   clen    = (const int*)d_in[29];

  float* out = (float*)d_out;
  char* ws = (char*)d_ws;
  size_t off = 0;
  auto alloc = [&](size_t bytes) -> void* {
    off = (off + 255) & ~(size_t)255;
    void* p = ws + off;
    off += bytes;
    return p;
  };

  // bf16 weight copies (fc_W: 61MB f32 -> 30MB bf16, L2-resident across 25 steps)
  void* wWihF = alloc((size_t)G4 * En * 2);
  void* wWhhF = alloc((size_t)G4 * Hn * 2);
  void* wWihB = alloc((size_t)G4 * En * 2);
  void* wWhhB = alloc((size_t)G4 * Hn * 2);
  void* wF4   = alloc((size_t)Hn * 2 * Hn * 2);
  void* wImg  = alloc((size_t)Hn * Hn * 2);
  void* wEAtt = alloc((size_t)An * Hn * 2);
  void* wDAtt = alloc((size_t)An * Hn * 2);
  void* wCWih = alloc((size_t)G4 * 2 * En * 2);
  void* wCWhh = alloc((size_t)G4 * Hn * 2);
  void* wFc   = alloc((size_t)Vn * Hn * 2);
  // activations
  void* embArt = alloc((size_t)ART * Bn * En * 2);         // bf16 [t][b][e]
  float* gih   = (float*)alloc((size_t)ART * Bn * G4 * 4); // f32 gates, reused per dir
  void* hsF    = alloc((size_t)ART * Bn * Hn * 2);
  void* hsB    = alloc((size_t)ART * Bn * Hn * 2);
  void* X2     = alloc((size_t)Bn * ART * 1024 * 2);
  void* h1     = alloc((size_t)Bn * ART * Hn * 2);
  void* encBf  = alloc((size_t)Bn * NP * Hn * 2);
  void* enc1   = alloc((size_t)Bn * NP * Hn * 2);
  void* hidden = alloc((size_t)Bn * Lc * Hn * 2);
  float* att1  = (float*)alloc((size_t)Bn * Lc * An * 4);
  float* cEnc  = (float*)alloc((size_t)Bn * Hn * 4);
  unsigned short* hBfE = (unsigned short*)alloc((size_t)Bn * Hn * 2);
  float* hDec  = (float*)alloc((size_t)Bn * Hn * 4);
  float* cDec  = (float*)alloc((size_t)Bn * Hn * 4);
  unsigned short* hBfD = (unsigned short*)alloc((size_t)Bn * Hn * 2);
  float* att2  = (float*)alloc((size_t)Bn * An * 4);
  float* eBuf  = (float*)alloc((size_t)Bn * Lc * 4);
  float* aBuf  = (float*)alloc((size_t)Bn * Lc * 4);
  unsigned short* xBf = (unsigned short*)alloc((size_t)Bn * 1024 * 2);
  float* gates = (float*)alloc((size_t)Bn * G4 * 4);
  float* logit = (float*)alloc((size_t)Bn * Vn * 4);

  auto cvt = [&](const float* s, void* d, size_t n) {
    k_f2bf<<<dim3((unsigned)((n + 255) / 256)), 256, 0, stream>>>(s, (unsigned short*)d, (int)n);
  };
  cvt(WihF, wWihF, (size_t)G4 * En);   cvt(WhhF, wWhhF, (size_t)G4 * Hn);
  cvt(WihB, wWihB, (size_t)G4 * En);   cvt(WhhB, wWhhB, (size_t)G4 * Hn);
  cvt(f4W, wF4, (size_t)Hn * 2 * Hn);  cvt(imgW, wImg, (size_t)Hn * Hn);
  cvt(encAttW, wEAtt, (size_t)An * Hn); cvt(decAttW, wDAtt, (size_t)An * Hn);
  cvt(cWih, wCWih, (size_t)G4 * 2 * En); cvt(cWhh, wCWhh, (size_t)G4 * Hn);
  cvt(fcW, wFc, (size_t)Vn * Hn);
  cvt(encoder_out, encBf, (size_t)Bn * NP * Hn);

  k_gather_art<<<dim3((ART * Bn * En + 255) / 256), 256, 0, stream>>>(arts, emb1_w, (unsigned short*)embArt);

  // ---- biLSTM encoder: hoist x@Wih^T for all 180 steps into one GEMM per dir ----
  const int MT = ART * Bn;  // 11520
  // forward
  GEMM_CALL(true, false, false, false, embArt, wWihF, bihF, gih, nullptr, MT, G4, En);
  k_zero_f32<<<dim3((Bn * Hn + 255) / 256), 256, 0, stream>>>(cEnc, Bn * Hn);
  k_zero_u16<<<dim3((Bn * Hn + 255) / 256), 256, 0, stream>>>(hBfE, Bn * Hn);
  for (int t = 0; t < ART; ++t) {
    float* gslice = gih + (size_t)t * Bn * G4;
    GEMM_CALL(true, true, false, false, hBfE, wWhhF, bhhF, gslice, nullptr, Bn, G4, Hn);
    k_lstm_step_enc<<<dim3((Bn * Hn + 255) / 256), 256, 0, stream>>>(
        gslice, cEnc, hBfE, (unsigned short*)hsF + (size_t)t * Bn * Hn);
  }
  // backward (consume timesteps in reverse, write hs at original index)
  GEMM_CALL(true, false, false, false, embArt, wWihB, bihB, gih, nullptr, MT, G4, En);
  k_zero_f32<<<dim3((Bn * Hn + 255) / 256), 256, 0, stream>>>(cEnc, Bn * Hn);
  k_zero_u16<<<dim3((Bn * Hn + 255) / 256), 256, 0, stream>>>(hBfE, Bn * Hn);
  for (int s = 0; s < ART; ++s) {
    int t = ART - 1 - s;
    float* gslice = gih + (size_t)t * Bn * G4;
    GEMM_CALL(true, true, false, false, hBfE, wWhhB, bhhB, gslice, nullptr, Bn, G4, Hn);
    k_lstm_step_enc<<<dim3((Bn * Hn + 255) / 256), 256, 0, stream>>>(
        gslice, cEnc, hBfE, (unsigned short*)hsB + (size_t)t * Bn * Hn);
  }

  // ---- f4 / image projections, hiddens, precomputed att1 ----
  k_concat_fb<<<dim3((Bn * ART * 1024 + 255) / 256), 256, 0, stream>>>(
      (const unsigned short*)hsF, (const unsigned short*)hsB, (unsigned short*)X2);
  GEMM_CALL(true, false, true, true, X2, wF4, f4b, nullptr, h1, Bn * ART, Hn, 2 * Hn);
  GEMM_CALL(true, false, true, true, encBf, wImg, imgb, nullptr, enc1, Bn * NP, Hn, Hn);
  k_assemble_hiddens<<<dim3((Bn * Lc * Hn + 255) / 256), 256, 0, stream>>>(
      (const unsigned short*)h1, (const unsigned short*)enc1, (unsigned short*)hidden);
  GEMM_CALL(true, false, false, false, hidden, wEAtt, encAttb, att1, nullptr, Bn * Lc, An, Hn);

  // ---- decoder ----
  k_zero_f32<<<dim3((Bn * Hn + 255) / 256), 256, 0, stream>>>(hDec, Bn * Hn);
  k_zero_f32<<<dim3((Bn * Hn + 255) / 256), 256, 0, stream>>>(cDec, Bn * Hn);
  k_zero_u16<<<dim3((Bn * Hn + 255) / 256), 256, 0, stream>>>(hBfD, Bn * Hn);

  for (int t = 0; t < TMAXn; ++t) {
    GEMM_CALL(true, false, false, false, hBfD, wDAtt, decAttb, att2, nullptr, Bn, An, Hn);
    k_att_scores<<<dim3((Bn * Lc + 7) / 8), 256, 0, stream>>>(att1, att2, fullW, fullb, eBuf);
    k_softmax_L<<<dim3(Bn), 256, 0, stream>>>(eBuf, aBuf);
    k_awe_x<<<dim3(Bn), 256, 0, stream>>>(aBuf, (const unsigned short*)hidden, xBf);
    k_emb_x<<<dim3((Bn * En + 255) / 256), 256, 0, stream>>>(caps, emb_w, xBf, t);
    GEMM_CALL(true, false, false, false, xBf, wCWih, cbih, gates, nullptr, Bn, G4, 2 * En);
    GEMM_CALL(true, true, false, false, hBfD, wCWhh, cbhh, gates, nullptr, Bn, G4, Hn);
    k_cell_step<<<dim3((Bn * Hn + 255) / 256), 256, 0, stream>>>(gates, clen, hDec, cDec, hBfD, t);
    GEMM_CALL(true, false, false, false, hBfD, wFc, fcb, logit, nullptr, Bn, Vn, Hn);
    k_log_softmax<<<dim3(Bn), 256, 0, stream>>>(logit, clen, out, t);
  }

  // ---- remaining tuple outputs: caps, decode_lengths, zero alphas/betas ----
  const size_t OFF_CAPS = (size_t)Bn * TMAXn * Vn;
  const size_t OFF_DLEN = OFF_CAPS + (size_t)Bn * (TMAXn + 1);
  const size_t OFF_AB   = OFF_DLEN + (size_t)Bn;
  const int N_AB = Bn * TMAXn * NP + Bn * TMAXn * 180;
  k_misc_out<<<dim3((Bn * (TMAXn + 1) + 255) / 256), 256, 0, stream>>>(
      caps, clen, out + OFF_CAPS, out + OFF_DLEN);
  k_zero_f32<<<dim3((N_AB + 255) / 256), 256, 0, stream>>>(out + OFF_AB, N_AB);
}
